// RNNAttention_5368709120365
// MI455X (gfx1250) — matrix operand order repaired and checked by reading, not compiled, not run on hardware
//
#include <hip/hip_runtime.h>

// ---------------------------------------------------------------------------
// RNN-with-attention for MI455X (gfx1250, wave32, WMMA, TDM).
//
// Shapes: B=16, T=128, M=512, d=u=256, gates=1024, Kcat=768.
// Strategy:
//   * keys GEMM (8192x256x256) once with v_wmma_f32_16x16x32_bf16; A tile
//     staged into LDS by the Tensor Data Mover (tensor_load_to_lds +
//     s_wait_tensorcnt), f32->bf16 converted in LDS.
//   * per step: attn_step (16 WGs, hw tanh + wave32 shuffle reductions),
//               lstm_step (1 WG, WMMA LSTM GEMM with register-resident gates,
//               fused next-step q GEMM).
//   * weights pre-packed into WMMA B-operand lane layout (bf16).
// ---------------------------------------------------------------------------

typedef __attribute__((ext_vector_type(16))) __bf16       v16bf;
typedef __attribute__((ext_vector_type(8)))  float        v8f;
typedef __attribute__((ext_vector_type(4)))  unsigned int u32x4;
typedef __attribute__((ext_vector_type(8)))  int          i32x8;
typedef __attribute__((ext_vector_type(4)))  int          i32x4;

union ABFrag {
    struct { u32x4 lo, hi; } q;   // 8 dwords = 16 bf16 = one WMMA operand
    v16bf v;
};

#define LOG2E 1.4426950408889634f

#if defined(__HIP_DEVICE_COMPILE__) && \
    __has_builtin(__builtin_amdgcn_tensor_load_to_lds) && \
    __has_builtin(__builtin_amdgcn_s_wait_tensorcnt)
#define USE_TDM 1
#else
#define USE_TDM 0
#endif

__device__ __forceinline__ float hw_tanh(float x) {
#if __has_builtin(__builtin_amdgcn_tanhf)
    return __builtin_amdgcn_tanhf(x);                 // v_tanh_f32 (TRANS unit)
#else
    float e = __builtin_amdgcn_exp2f(2.0f * LOG2E * x);
    return (e - 1.0f) * __builtin_amdgcn_rcpf(e + 1.0f);
#endif
}

__device__ __forceinline__ float hw_sigmoid(float x) {
    return __builtin_amdgcn_rcpf(1.0f + __builtin_amdgcn_exp2f(-LOG2E * x));
}

__device__ __forceinline__ v8f wmma_bf16(v16bf a, v16bf b, v8f c) {
    // D = A(16x32 bf16) x B(32x16 bf16) + C(16x16 f32)
    return __builtin_amdgcn_wmma_f32_16x16x32_bf16(
        false, a, false, b, (short)0, c, false, false);
}

// ---------------------------------------------------------------------------
// Prep 1: pack weights (f32 row-major, K x N) into WMMA B-operand layout.
// For a (tile nt, kstep ks) block of 512 bf16: lane l holds col n = nt*16+(l&15),
// dword v holds K pair k = ks*32 + (v&4)*4 + (l>>4)*8 + 2*(v&3) (+0/+1 in lo/hi half).
// Jobs: 0..1535  -> Wcat = [lstm_W ; lstm_U] (768x1024), 64 tiles x 24 ksteps
//       1536..1663 -> key_W   (256x256), 16 x 8
//       1664..1791 -> query_W (256x256), 16 x 8
// ---------------------------------------------------------------------------
__global__ __launch_bounds__(256) void pack_weights(
    const float* __restrict__ lstm_W, const float* __restrict__ lstm_U,
    const float* __restrict__ key_W,  const float* __restrict__ query_W,
    __bf16* __restrict__ wcat_p, __bf16* __restrict__ keyW_p,
    __bf16* __restrict__ qW_p)
{
    int job = blockIdx.x, tid = threadIdx.x;
    int lane = tid >> 3, v = tid & 7;
    int n_l = lane & 15, hf = lane >> 4;

    float w0, w1;
    unsigned int* dst;
    if (job < 1536) {
        int nt = job / 24, ks = job % 24;
        int k = ks * 32 + (v & 4) * 4 + hf * 8 + (v & 3) * 2;
        int n = nt * 16 + n_l;
        w0 = (k     < 512) ? lstm_W[k * 1024 + n]       : lstm_U[(k - 512) * 1024 + n];
        w1 = (k + 1 < 512) ? lstm_W[(k + 1) * 1024 + n] : lstm_U[(k - 511) * 1024 + n];
        dst = (unsigned int*)wcat_p + job * 256;
    } else if (job < 1664) {
        int j = job - 1536;
        int nt = j / 8, ks = j % 8;
        int k = ks * 32 + (v & 4) * 4 + hf * 8 + (v & 3) * 2;
        int n = nt * 16 + n_l;
        w0 = key_W[k * 256 + n];
        w1 = key_W[(k + 1) * 256 + n];
        dst = (unsigned int*)keyW_p + j * 256;
    } else {
        int j = job - 1664;
        int nt = j / 8, ks = j % 8;
        int k = ks * 32 + (v & 4) * 4 + hf * 8 + (v & 3) * 2;
        int n = nt * 16 + n_l;
        w0 = query_W[k * 256 + n];
        w1 = query_W[(k + 1) * 256 + n];
        dst = (unsigned int*)qW_p + j * 256;
    }
    unsigned short b0 = __builtin_bit_cast(unsigned short, (__bf16)w0);
    unsigned short b1 = __builtin_bit_cast(unsigned short, (__bf16)w1);
    dst[lane * 8 + v] = (unsigned int)b0 | ((unsigned int)b1 << 16);
}

// Prep 2: inputs (B,T,d) f32 -> bf16.
__global__ __launch_bounds__(256) void cast_inputs(
    const float* __restrict__ x, __bf16* __restrict__ xb, int n)
{
    int i = blockIdx.x * 256 + threadIdx.x;
    if (i < n) xb[i] = (__bf16)x[i];
}

// Prep 3: h0=0, c0=0, q0 = 0 @ query_W + query_b = query_b.
__global__ __launch_bounds__(256) void init_state(
    const float* __restrict__ query_b, float* __restrict__ q_buf,
    float* __restrict__ c_ws, __bf16* __restrict__ h_bg)
{
    int b = blockIdx.x, n = threadIdx.x;
    q_buf[b * 256 + n] = query_b[n];
    c_ws[b * 256 + n]  = 0.0f;
    h_bg[b * 256 + n]  = (__bf16)0.0f;
}

// ---------------------------------------------------------------------------
// Prep 4: keys = attended(8192x256) @ key_W(256x256) + key_b, WMMA bf16.
// One WG per 16-row M tile (the tile is a CONTIGUOUS 16 KB block, so it is a
// trivial 1-D TDM tile): wave 0 issues one tensor_load_to_lds descriptor,
// waits on TENSORcnt, then all 256 threads convert f32 -> bf16 in LDS.
// 8 waves each own 2 N tiles; A fragment loaded once per k-step (hoisted).
// ---------------------------------------------------------------------------
__global__ __launch_bounds__(256) void keys_gemm(
    const float* __restrict__ attended, const __bf16* __restrict__ keyW_p,
    const float* __restrict__ key_b, float* __restrict__ keys)
{
    __shared__ __align__(16) float  af_sh[16 * 256];   // 16 KB TDM destination
    __shared__ __align__(16) __bf16 a_sh[16 * 256];    //  8 KB WMMA A tile
    int mt = blockIdx.x, tid = threadIdx.x;
    const float* asrc = attended + (size_t)mt * 16 * 256;

#if USE_TDM
    if (tid < 32) {   // one descriptor per WG; TDM ignores EXEC
        unsigned long long ga = (unsigned long long)asrc;
        // generic LDS pointer: addr[31:0] is the LDS byte offset (ISA 10.2)
        unsigned lds_base = (unsigned)(unsigned long long)(void*)af_sh;
        u32x4 g0 = { 1u,                                    // count=1, user D#
                     lds_base,                              // lds_addr
                     (unsigned)ga,                          // global_addr[31:0]
                     ((unsigned)(ga >> 32) & 0x01FFFFFFu)   // global_addr[56:32]
                     | (2u << 30) };                        // type = 2 (image)
        i32x8 g1 = { 0x00020000,   // data_size=4B, no multicast/pad/iterate
                     0x10000000,   // tensor_dim0 = 4096 elements (bits[63:48])
                     0,            // tensor_dim0 hi / tensor_dim1 lo = 0
                     0x10000000,   // tile_dim0 = 4096 (bits[127:112])
                     0,            // tile_dim1 = tile_dim2 = 0 (1-D tile)
                     4096,         // tensor_dim0_stride (bits[191:160])
                     0, 0 };
        i32x4 gz4 = { 0, 0, 0, 0 };            // groups 2/3 unused (<=2-D)
        i32x8 gz8 = { 0, 0, 0, 0, 0, 0, 0, 0 }; // trailing group (6-arg form)
        __builtin_amdgcn_tensor_load_to_lds(g0, g1, gz4, gz4, gz8, 0);
        __builtin_amdgcn_s_wait_tensorcnt(0);
    }
    __syncthreads();
#pragma unroll
    for (int i = 0; i < 16; ++i) {
        int f = i * 256 + tid;
        a_sh[f] = (__bf16)af_sh[f];
    }
#else
#pragma unroll
    for (int i = 0; i < 16; ++i) {
        int f = i * 256 + tid;
        a_sh[f] = (__bf16)asrc[f];
    }
#endif
    __syncthreads();

    int wv = tid >> 5, lane = tid & 31, hf = lane >> 4, lo16 = lane & 15;
    const unsigned int* ash_dw = (const unsigned int*)a_sh;

    v8f z8 = {0, 0, 0, 0, 0, 0, 0, 0};
    v8f acc[2] = {z8, z8};
    for (int ks = 0; ks < 8; ++ks) {
        ABFrag a;
        int ad = lo16 * 128 + ks * 16 + hf * 4;        // dword index in A row
        a.q.lo = *(const u32x4*)(ash_dw + ad);
        a.q.hi = *(const u32x4*)(ash_dw + ad + 8);
#pragma unroll
        for (int tt = 0; tt < 2; ++tt) {
            int nt = wv * 2 + tt;
            const unsigned int* bp = (const unsigned int*)keyW_p
                + ((size_t)(nt * 8 + ks) * 32 + lane) * 8;
            ABFrag b;
            b.q.lo = *(const u32x4*)bp;
            b.q.hi = *(const u32x4*)(bp + 4);
            acc[tt] = wmma_bf16(a.v, b.v, acc[tt]);
        }
    }
#pragma unroll
    for (int tt = 0; tt < 2; ++tt) {
        int nt = wv * 2 + tt;
        int n = nt * 16 + lo16;
        float bias = key_b[n];
        float* orow = keys + (size_t)mt * 16 * 256 + n;
#pragma unroll
        for (int r = 0; r < 8; ++r)
            orow[(r + 8 * hf) * 256] = acc[tt][r] + bias;  // C: M=r+8*hf, N=lo16
    }
}

// ---------------------------------------------------------------------------
// Per step: attention. One WG per batch (grid=16, 256 threads = 8 waves).
// scores[m] = sum_u tanh(keys[b,m,u] + q[b,u]) * aw[u]; softmax; ctx GEMV.
// ---------------------------------------------------------------------------
__global__ __launch_bounds__(256) void attn_step(
    const float* __restrict__ keys, const float* __restrict__ q_buf,
    const float* __restrict__ attn_W, const float* __restrict__ attn_b,
    const float* __restrict__ attended, __bf16* __restrict__ ctx_b)
{
    int b = blockIdx.x, tid = threadIdx.x;
    __shared__ float q_s[256], aw_s[256], red[256];
    __shared__ float sc[512];

    q_s[tid]  = q_buf[b * 256 + tid];
    aw_s[tid] = attn_W[tid];
    __syncthreads();

    int wv = tid >> 5, lane = tid & 31;
    float ab0 = attn_b[0];
    const float* kb = keys + (size_t)b * 512 * 256;

    // wave wv handles rows m = wv, wv+8, ... (64 rows each)
    for (int i = 0; i < 64; ++i) {
        int m = wv + (i << 3);
        const float* krow = kb + m * 256;
        float acc = 0.0f;
#pragma unroll
        for (int uc = 0; uc < 8; ++uc) {
            int u = uc * 32 + lane;
            acc += hw_tanh(krow[u] + q_s[u]) * aw_s[u];
        }
#pragma unroll
        for (int o = 16; o; o >>= 1) acc += __shfl_xor(acc, o, 32);
        if (lane == 0) sc[m] = acc + ab0;
    }
    __syncthreads();

    // softmax over 512
    float mx = fmaxf(sc[tid], sc[tid + 256]);
    red[tid] = mx; __syncthreads();
    for (int s = 128; s; s >>= 1) {
        if (tid < s) red[tid] = fmaxf(red[tid], red[tid + s]);
        __syncthreads();
    }
    mx = red[0];
    __syncthreads();
    float e0 = __builtin_amdgcn_exp2f(LOG2E * (sc[tid] - mx));
    float e1 = __builtin_amdgcn_exp2f(LOG2E * (sc[tid + 256] - mx));
    red[tid] = e0 + e1; __syncthreads();
    for (int s = 128; s; s >>= 1) {
        if (tid < s) red[tid] += red[tid + s];
        __syncthreads();
    }
    float inv = __builtin_amdgcn_rcpf(red[0]);
    __syncthreads();
    sc[tid]       = e0 * inv;
    sc[tid + 256] = e1 * inv;
    __syncthreads();

    // ctx[b, tid] = sum_m a[m] * attended[b, m, tid]
    const float* arow = attended + (size_t)b * 512 * 256 + tid;
    float acc = 0.0f;
#pragma unroll 8
    for (int m = 0; m < 512; ++m) acc += sc[m] * arow[m * 256];
    ctx_b[b * 256 + tid] = (__bf16)acc;
}

// ---------------------------------------------------------------------------
// Per step: LSTM GEMM + gates + next-step q GEMM. One WG of 512 thr (16 waves).
// z = [x_t | ctx | h] (16x768 bf16, LDS) @ Wcat_pack (768x1024 bf16).
// Wave wv owns n in [wv*16, wv*16+16) across all 4 gates (tiles wv, wv+16,
// wv+32, wv+48) -> gate fusion entirely in registers, no z scratch.
// ---------------------------------------------------------------------------
__global__ __launch_bounds__(512) void lstm_step(
    const __bf16* __restrict__ x_b, const __bf16* __restrict__ ctx_b,
    __bf16* __restrict__ h_bg, const __bf16* __restrict__ wcat_p,
    const float* __restrict__ lstm_b, float* __restrict__ c_ws,
    const __bf16* __restrict__ qW_p, const float* __restrict__ query_b,
    float* __restrict__ q_buf, float* __restrict__ seq_out,
    float* __restrict__ hT_out, float* __restrict__ cT_out, int t)
{
    __shared__ __align__(16) __bf16 a_sh[16 * 768];   // 24 KB
    __shared__ __align__(16) __bf16 h_sh[16 * 256];   //  8 KB
    int tid = threadIdx.x;

    // stage A = [x_t | ctx | h] as bf16 dwords
    unsigned int* asd = (unsigned int*)a_sh;
    const unsigned int* xdw = (const unsigned int*)x_b;
    const unsigned int* cdw = (const unsigned int*)ctx_b;
    const unsigned int* hdw = (const unsigned int*)h_bg;
#pragma unroll
    for (int i = 0; i < 12; ++i) {
        int f = i * 512 + tid;                 // 0..6143
        int row = f / 384, c = f - row * 384;  // row = batch, c = dword in row
        unsigned int val;
        if (c < 128)      val = xdw[(row * 128 + t) * 128 + c];
        else if (c < 256) val = cdw[row * 128 + (c - 128)];
        else              val = hdw[row * 128 + (c - 256)];
        asd[f] = val;
    }
    __syncthreads();

    int wv = tid >> 5, lane = tid & 31, hf = lane >> 4, lo16 = lane & 15;
    const unsigned int* ash_dw = (const unsigned int*)a_sh;

    v8f z8 = {0, 0, 0, 0, 0, 0, 0, 0};
    v8f acc[4] = {z8, z8, z8, z8};             // i, f, g, o tiles of same n-range

    for (int ks = 0; ks < 24; ++ks) {
        ABFrag a;
        int ad = lo16 * 384 + ks * 16 + hf * 4;
        a.q.lo = *(const u32x4*)(ash_dw + ad);
        a.q.hi = *(const u32x4*)(ash_dw + ad + 8);
#pragma unroll
        for (int g = 0; g < 4; ++g) {
            int nt = wv + g * 16;
            const unsigned int* bp =
                (const unsigned int*)wcat_p + ((size_t)(nt * 24 + ks) * 32 + lane) * 8;
            ABFrag bf;
            bf.q.lo = *(const u32x4*)bp;
            bf.q.hi = *(const u32x4*)(bp + 4);
            if (ks < 23) __builtin_prefetch(bp + 256, 0, 1);   // next ks block
            acc[g] = wmma_bf16(a.v, bf.v, acc[g]);
        }
    }

    // gates: element (b = r + 8*hf, n = wv*16 + lo16)
    int n = wv * 16 + lo16;
    float bi = lstm_b[n], bf_ = lstm_b[256 + n];
    float bg = lstm_b[512 + n], bo = lstm_b[768 + n];
#pragma unroll
    for (int r = 0; r < 8; ++r) {
        int bidx = r + 8 * hf;
        float ig = hw_sigmoid(acc[0][r] + bi);
        float fg = hw_sigmoid(acc[1][r] + bf_);
        float gg = hw_tanh   (acc[2][r] + bg);
        float og = hw_sigmoid(acc[3][r] + bo);
        float cold = c_ws[bidx * 256 + n];
        float cn = fg * cold + ig * gg;
        float hn = og * hw_tanh(cn);
        c_ws[bidx * 256 + n] = cn;
        seq_out[((size_t)bidx * 128 + t) * 256 + n] = hn;
        __bf16 hb = (__bf16)hn;
        h_bg[bidx * 256 + n] = hb;
        h_sh[bidx * 256 + n] = hb;
        if (t == 127) {
            hT_out[bidx * 256 + n] = hn;
            cT_out[bidx * 256 + n] = cn;
        }
    }
    __syncthreads();

    // q = h_new @ query_W + query_b (for next step); wave wv owns N tile wv
    const unsigned int* hsd = (const unsigned int*)h_sh;
    v8f qa = {0, 0, 0, 0, 0, 0, 0, 0};
    for (int ks = 0; ks < 8; ++ks) {
        ABFrag a, b;
        int ad = lo16 * 128 + ks * 16 + hf * 4;
        a.q.lo = *(const u32x4*)(hsd + ad);
        a.q.hi = *(const u32x4*)(hsd + ad + 8);
        const unsigned int* bp =
            (const unsigned int*)qW_p + ((size_t)(wv * 8 + ks) * 32 + lane) * 8;
        b.q.lo = *(const u32x4*)bp;
        b.q.hi = *(const u32x4*)(bp + 4);
        qa = wmma_bf16(a.v, b.v, qa);
    }
    float qb = query_b[n];
#pragma unroll
    for (int r = 0; r < 8; ++r)
        q_buf[(r + 8 * hf) * 256 + n] = qa[r] + qb;
}

// ---------------------------------------------------------------------------
extern "C" void kernel_launch(void* const* d_in, const int* in_sizes, int n_in,
                              void* d_out, int out_size, void* d_ws, size_t ws_size,
                              hipStream_t stream)
{
    const float* inputs   = (const float*)d_in[0];
    const float* attended = (const float*)d_in[1];
    const float* key_W    = (const float*)d_in[2];
    const float* key_b    = (const float*)d_in[3];
    const float* query_W  = (const float*)d_in[4];
    const float* query_b  = (const float*)d_in[5];
    const float* attn_W   = (const float*)d_in[6];
    const float* attn_b   = (const float*)d_in[7];
    const float* lstm_W   = (const float*)d_in[8];
    const float* lstm_U   = (const float*)d_in[9];
    const float* lstm_b   = (const float*)d_in[10];
    float* out = (float*)d_out;

    char* ws = (char*)d_ws;
    size_t off = 0;
    auto take = [&](size_t bytes) -> char* {
        char* p = ws + off;
        off += (bytes + 255) & ~(size_t)255;
        return p;
    };
    float*  keys   = (float*) take((size_t)16 * 512 * 256 * 4);  // 8 MB
    __bf16* x_b    = (__bf16*)take((size_t)16 * 128 * 256 * 2);  // 1 MB
    __bf16* wcat_p = (__bf16*)take((size_t)64 * 24 * 512 * 2);   // 1.5 MB
    __bf16* keyW_p = (__bf16*)take((size_t)16 * 8 * 512 * 2);
    __bf16* qW_p   = (__bf16*)take((size_t)16 * 8 * 512 * 2);
    float*  q_buf  = (float*) take((size_t)16 * 256 * 4);
    __bf16* ctx_b  = (__bf16*)take((size_t)16 * 256 * 2);
    __bf16* h_bg   = (__bf16*)take((size_t)16 * 256 * 2);
    float*  c_ws   = (float*) take((size_t)16 * 256 * 4);

    pack_weights<<<1792, 256, 0, stream>>>(lstm_W, lstm_U, key_W, query_W,
                                           wcat_p, keyW_p, qW_p);
    cast_inputs<<<2048, 256, 0, stream>>>(inputs, x_b, 16 * 128 * 256);
    init_state<<<16, 256, 0, stream>>>(query_b, q_buf, c_ws, h_bg);
    keys_gemm<<<512, 256, 0, stream>>>(attended, keyW_p, key_b, keys);

    float* seq = out;
    float* hT  = out + (size_t)16 * 128 * 256;
    float* cT  = hT + 16 * 256;

    for (int t = 0; t < 128; ++t) {
        attn_step<<<16, 256, 0, stream>>>(keys, q_buf, attn_W, attn_b,
                                          attended, ctx_b);
        lstm_step<<<1, 512, 0, stream>>>(x_b, ctx_b, h_bg, wcat_p, lstm_b,
                                         c_ws, qW_p, query_b, q_buf,
                                         seq, hT, cT, t);
    }
}
